// EdgeAdditionLayer_82162724372846
// MI455X (gfx1250) — compile-verified, matrix operand order checked by reading, not run on hardware
//
#include <hip/hip_runtime.h>
#include <hip/hip_bf16.h>
#include <math.h>

typedef __attribute__((ext_vector_type(16))) _Float16 v16h;
typedef __attribute__((ext_vector_type(8)))  _Float16 v8h;
typedef __attribute__((ext_vector_type(4)))  _Float16 v4h;
typedef __attribute__((ext_vector_type(8)))  float    v8f;

#define NN   1536
#define DD   256
#define HH   64
#define EDIM 32
#define KK   16
#define NE   24576

// ---------------------------------------------------------------------------
// DPP 16-lane rotate-reduce step (row_ror:N), pure VALU — no LDS traffic.
// Rotating by 8,4,2,1 within each 16-lane row leaves every lane holding the
// sum of its row.
// ---------------------------------------------------------------------------
template <int N>
__device__ __forceinline__ float rotadd(float v) {
    int m = __builtin_amdgcn_update_dpp(0, __builtin_bit_cast(int, v),
                                        0x120 + N /*ROW_ROR:N*/, 0xF, 0xF, false);
    return v + __builtin_bit_cast(float, m);
}

// ---------------------------------------------------------------------------
// Kernel 0a: zero the duplicate-edge bitmap (must happen every launch)
// ---------------------------------------------------------------------------
__global__ void zero_bits_kernel(unsigned int* __restrict__ p, int n) {
    int t = blockIdx.x * blockDim.x + threadIdx.x;
    if (t < n) p[t] = 0u;
}

// ---------------------------------------------------------------------------
// Kernel 0b: scatter existing edges into an N*N bitmap
// ---------------------------------------------------------------------------
__global__ void scatter_edges_kernel(const long long* __restrict__ ei,
                                     unsigned int* __restrict__ bits) {
    int e = blockIdx.x * blockDim.x + threadIdx.x;
    if (e < NE) {
        int s  = (int)ei[e];
        int tg = (int)ei[NE + e];
        unsigned int pos = (unsigned int)(s * NN + tg);
        atomicOr(&bits[pos >> 5], 1u << (pos & 31));
    }
}

// ---------------------------------------------------------------------------
// Kernel 1: four projections  A = X@W1[:,:D]^T, B = X@W1[:,D:]^T,
//                             AE = X@We1[:,:D]^T, BE = X@We1[:,D:]^T
// ---------------------------------------------------------------------------
__global__ void proj_kernel(const float* __restrict__ X,
                            const float* __restrict__ W1,
                            const float* __restrict__ We1,
                            float* __restrict__ A,  float* __restrict__ B,
                            float* __restrict__ AE, float* __restrict__ BE) {
    __shared__ float xs[DD];
    const int i = blockIdx.x;
    const int t = threadIdx.x;
    xs[t] = X[i * DD + t];
    __syncthreads();
    const int sel = t >> 6;   // 0..3
    const int h   = t & 63;
    const float* w = (sel < 2 ? W1 : We1) + h * (2 * DD) + (sel & 1) * DD;
    float acc = 0.f;
    #pragma unroll 8
    for (int d = 0; d < DD; ++d) acc += xs[d] * w[d];
    float* dst = (sel == 0) ? A : (sel == 1) ? B : (sel == 2) ? AE : BE;
    dst[i * HH + h] = acc;
}

// ---------------------------------------------------------------------------
// Kernel 2: fused similarity MLP -> sim[N][N] via WMMA f16 16x16x32.
// Wave w of block b owns row i = b*8+w; loops over 96 blocks of 16 columns.
// h1 fragments built with packed f16 math from an f16-staged B tile;
// epilogue reduction via DPP row-rotates (no LDS).
// ---------------------------------------------------------------------------
__device__ __forceinline__ v8f wmma_f16(v16h a, v16h b, v8f c) {
    return __builtin_amdgcn_wmma_f32_16x16x32_f16(false, a, false, b,
                                                  (short)0, c, false, false);
}

__global__ void sim_kernel(const float* __restrict__ Amat,
                           const float* __restrict__ Bmat,
                           const float* __restrict__ b1,
                           const float* __restrict__ W2,
                           const float* __restrict__ b2,
                           const float* __restrict__ W3,
                           const float* __restrict__ b3,
                           float* __restrict__ sim) {
    __shared__ __align__(16) _Float16 Bsh[16 * HH];   // 16 B-rows as f16 (2 KB)
    const int tid  = threadIdx.x;
    const int lane = tid & 31;
    const int wv   = tid >> 5;                 // wave 0..7
    const int nl   = lane & 15;
    const int half = lane >> 4;
    const int i    = blockIdx.x * 8 + wv;      // sim row this wave owns

    // Loop-invariant W2 B-fragments: wb[ntile][kstep]; B[k,n] = W2[n,k].
    v16h wb00, wb01, wb10, wb11;
    #pragma unroll
    for (int e = 0; e < 16; ++e) {
        int kb = ((e & 8) << 1) + half * 8 + (e & 7);   // k within 32-chunk
        wb00[e] = (_Float16)W2[(nl     ) * HH + kb];
        wb01[e] = (_Float16)W2[(nl     ) * HH + 32 + kb];
        wb10[e] = (_Float16)W2[(16 + nl) * HH + kb];
        wb11[e] = (_Float16)W2[(16 + nl) * HH + 32 + kb];
    }

    // Per-lane A' = f16(A[i,k]+b1[k]) packed for this lane's k runs.
    // Fragment element e<8  -> k = ks*32 + half*8 + e        (alo)
    // Fragment element e>=8 -> k = ks*32 + half*8 + 16 + e-8 (ahi)
    v8h alo[2], ahi[2];
    #pragma unroll
    for (int ks = 0; ks < 2; ++ks) {
        #pragma unroll
        for (int e = 0; e < 8; ++e) {
            int k0 = ks * 32 + half * 8 + e;
            int k1 = k0 + 16;
            alo[ks][e] = (_Float16)(Amat[i * HH + k0] + b1[k0]);
            ahi[ks][e] = (_Float16)(Amat[i * HH + k1] + b1[k1]);
        }
    }

    const float b2a = b2[nl],      b2b = b2[16 + nl];
    const float w3a = W3[nl],      w3b = W3[16 + nl];
    const float b3v = b3[0];

    for (int jb = 0; jb < NN / 16; ++jb) {
        const int j0 = jb * 16;
        // cooperative stage of B[j0..j0+15, 0:64] as f16 (1024 halves)
        {
            const float4* gsrc = (const float4*)(Bmat + j0 * HH);
            float4 f = gsrc[tid];
            v4h p = { (_Float16)f.x, (_Float16)f.y, (_Float16)f.z, (_Float16)f.w };
            ((v4h*)Bsh)[tid] = p;
        }
        __syncthreads();

        // Build A-fragments for h1 (M = j-row = nl) with packed f16 math.
        const _Float16* browh = Bsh + nl * HH;
        v16h afrag0, afrag1;
        {
            v8h z = {};
            v8h b00 = *(const v8h*)(browh + half * 8);
            v8h b01 = *(const v8h*)(browh + half * 8 + 16);
            v8h b10 = *(const v8h*)(browh + 32 + half * 8);
            v8h b11 = *(const v8h*)(browh + 32 + half * 8 + 16);
            v8h s00 = __builtin_elementwise_max(alo[0] + b00, z);
            v8h s01 = __builtin_elementwise_max(ahi[0] + b01, z);
            v8h s10 = __builtin_elementwise_max(alo[1] + b10, z);
            v8h s11 = __builtin_elementwise_max(ahi[1] + b11, z);
            afrag0 = __builtin_shufflevector(s00, s01, 0,1,2,3,4,5,6,7,
                                             8,9,10,11,12,13,14,15);
            afrag1 = __builtin_shufflevector(s10, s11, 0,1,2,3,4,5,6,7,
                                             8,9,10,11,12,13,14,15);
        }

        v8f c0 = {}; v8f c1 = {};
        c0 = wmma_f16(afrag0, wb00, c0);
        c0 = wmma_f16(afrag1, wb01, c0);
        c1 = wmma_f16(afrag0, wb10, c1);
        c1 = wmma_f16(afrag1, wb11, c1);

        // epilogue: relu(h2+b2) dot W3, reduced inside each 16-lane row (DPP)
        float part[8];
        #pragma unroll
        for (int r = 0; r < 8; ++r) {
            float v = fmaxf(c0[r] + b2a, 0.f) * w3a
                    + fmaxf(c1[r] + b2b, 0.f) * w3b;
            v = rotadd<8>(v);
            v = rotadd<4>(v);
            v = rotadd<2>(v);
            v = rotadd<1>(v);
            part[r] = v;
        }
        if (nl == 0) {   // lane 0 -> rows j0..j0+7, lane 16 -> rows j0+8..j0+15
            #pragma unroll
            for (int r = 0; r < 8; ++r) {
                float x = part[r] + b3v;
                sim[i * NN + j0 + half * 8 + r] = 1.f / (1.f + __expf(-x));
            }
        }
        __syncthreads();
    }
}

// ---------------------------------------------------------------------------
// Kernel 3: per-row top-K (K=16) with diagonal masking + thresh/dup check.
// ---------------------------------------------------------------------------
__global__ void topk_kernel(const float* __restrict__ sim,
                            const unsigned int* __restrict__ bits,
                            float* __restrict__ tkv,
                            int*   __restrict__ tki,
                            int*   __restrict__ tkvalid) {
    __shared__ float row[NN];
    __shared__ float sv[256];
    __shared__ int   si[256];
    const int i = blockIdx.x;
    const int t = threadIdx.x;
    #pragma unroll
    for (int q = 0; q < NN / 256; ++q) {
        int j = t + q * 256;
        row[j] = (j == i) ? -1.f : sim[i * NN + j];
    }
    __syncthreads();
    for (int kk = 0; kk < KK; ++kk) {
        float bv = -1e30f; int bj = NN;
        #pragma unroll
        for (int q = 0; q < NN / 256; ++q) {
            int j = t + q * 256;
            float v = row[j];
            if (v > bv || (v == bv && j < bj)) { bv = v; bj = j; }
        }
        sv[t] = bv; si[t] = bj;
        __syncthreads();
        for (int s = 128; s > 0; s >>= 1) {
            if (t < s) {
                float ov = sv[t + s]; int oj = si[t + s];
                if (ov > sv[t] || (ov == sv[t] && oj < si[t])) {
                    sv[t] = ov; si[t] = oj;
                }
            }
            __syncthreads();
        }
        if (t == 0) {
            int   j = si[0];
            float v = sv[0];
            tkv[i * KK + kk] = v;
            tki[i * KK + kk] = j;
            unsigned int pos = (unsigned int)(i * NN + j);
            int dup = (bits[pos >> 5] >> (pos & 31)) & 1;
            tkvalid[i * KK + kk] = (v >= 0.5f && !dup) ? 1 : 0;
            row[j] = -2.f;          // exclude from later rounds
        }
        __syncthreads();
    }
}

// ---------------------------------------------------------------------------
// Kernel 4: edge-feature MLP on selected pairs; invalid pairs -> zeros
// (reference multiplies ef by the 0/1 mask anyway).
// ---------------------------------------------------------------------------
__global__ void edge_kernel(const float* __restrict__ AE,
                            const float* __restrict__ BE,
                            const float* __restrict__ be1,
                            const float* __restrict__ We2,
                            const float* __restrict__ be2,
                            const float* __restrict__ tkv,
                            const int*   __restrict__ tki,
                            const int*   __restrict__ tkvalid,
                            float* __restrict__ out) {
    int p = blockIdx.x * blockDim.x + threadIdx.x;
    if (p >= NN * KK) return;
    const int i = p / KK;
    float* op = out + p * (EDIM + 1);
    if (!tkvalid[p]) {
        #pragma unroll
        for (int g = 0; g <= EDIM; ++g) op[g] = 0.f;
        return;
    }
    const int tg = tki[p];
    float e1[HH];
    #pragma unroll
    for (int h = 0; h < HH; ++h)
        e1[h] = fmaxf(AE[i * HH + h] + BE[tg * HH + h] + be1[h], 0.f);
    #pragma unroll 1
    for (int g = 0; g < EDIM; ++g) {
        float acc = be2[g];
        #pragma unroll
        for (int h = 0; h < HH; ++h) acc += e1[h] * We2[g * HH + h];
        op[g] = acc;
    }
    op[EDIM] = tkv[p];
}

// ---------------------------------------------------------------------------
extern "C" void kernel_launch(void* const* d_in, const int* in_sizes, int n_in,
                              void* d_out, int out_size, void* d_ws, size_t ws_size,
                              hipStream_t stream) {
    const float*     X   = (const float*)d_in[0];
    const long long* EI  = (const long long*)d_in[1];   // int64 edge_index [2, NE]
    const float*     W1  = (const float*)d_in[2];
    const float*     b1  = (const float*)d_in[3];
    const float*     W2  = (const float*)d_in[4];
    const float*     b2  = (const float*)d_in[5];
    const float*     W3  = (const float*)d_in[6];
    const float*     b3  = (const float*)d_in[7];
    const float*     We1 = (const float*)d_in[8];
    const float*     be1 = (const float*)d_in[9];
    const float*     We2 = (const float*)d_in[10];
    const float*     be2 = (const float*)d_in[11];
    float* out = (float*)d_out;

    char*  ws  = (char*)d_ws;
    size_t off = 0;
    auto carve = [&](size_t bytes) -> void* {
        void* p = ws + off;
        off += (bytes + 255) & ~(size_t)255;
        return p;
    };
    float*        A    = (float*)carve((size_t)NN * HH * 4);
    float*        B    = (float*)carve((size_t)NN * HH * 4);
    float*        AE   = (float*)carve((size_t)NN * HH * 4);
    float*        BE   = (float*)carve((size_t)NN * HH * 4);
    float*        simM = (float*)carve((size_t)NN * NN * 4);
    unsigned int* bits = (unsigned int*)carve((size_t)NN * NN / 8);
    float*        tkv  = (float*)carve((size_t)NN * KK * 4);
    int*          tki  = (int*)carve((size_t)NN * KK * 4);
    int*          tkva = (int*)carve((size_t)NN * KK * 4);

    const int nbits = NN * NN / 32;
    zero_bits_kernel<<<(nbits + 255) / 256, 256, 0, stream>>>(bits, nbits);
    scatter_edges_kernel<<<(NE + 255) / 256, 256, 0, stream>>>(EI, bits);
    proj_kernel<<<NN, 256, 0, stream>>>(X, W1, We1, A, B, AE, BE);
    sim_kernel<<<NN / 8, 256, 0, stream>>>(A, B, b1, W2, b2, W3, b3, simM);
    topk_kernel<<<NN, 256, 0, stream>>>(simM, bits, tkv, tki, tkva);
    edge_kernel<<<(NN * KK + 255) / 256, 256, 0, stream>>>(AE, BE, be1, We2, be2,
                                                           tkv, tki, tkva, out);
}
